// CGCNNRegressorStrong_64080912056812
// MI455X (gfx1250) — matrix operand order repaired and checked by reading, not compiled, not run on hardware
//
#include <hip/hip_runtime.h>
#include <hip/hip_bf16.h>
#include <math.h>

// ---------------- problem constants (match reference) ----------------
constexpr int N_   = 50000;
constexpr int E_   = 200000;
constexpr int H_   = 128;
constexpr int ED_  = 50;
constexpr int NG_  = 256;
constexpr int L_   = 6;
constexpr int Z_   = 2 * H_ + ED_; // 306

typedef __attribute__((ext_vector_type(16))) _Float16 v16h;
typedef __attribute__((ext_vector_type(8)))  _Float16 v8h;
typedef __attribute__((ext_vector_type(8)))  float    v8f;

// ---------------- device helpers ----------------
__device__ __forceinline__ float siluf(float x) { return x / (1.f + __expf(-x)); }
__device__ __forceinline__ float sigmoidf_(float x) { return 1.f / (1.f + __expf(-x)); }
__device__ __forceinline__ float softplusf_(float x) {
  return x > 0.f ? x + log1pf(__expf(-x)) : log1pf(__expf(x));
}
// p = [gamma(C), beta(C), mean(C), var(C)] rows
__device__ __forceinline__ float bnf(float x, const float* __restrict__ p, int C, int c) {
  return (x - p[2 * C + c]) * rsqrtf(p[3 * C + c] + 1e-5f) * p[0 * C + c] + p[1 * C + c];
}

// ---------------- WMMA GEMM: C[M,Nout] = A[M,K] * B[K,Nout] ----------------
// A row-major fp16; BT is B transposed (row-major [Nout,K]) fp16. One wave
// computes one 16x16 tile, accumulating fp32 via v_wmma_f32_16x16x32_f16.
// M, Nout multiples of 16; K multiple of 32.
template <typename OutT>
__global__ void k_gemm_wmma(const _Float16* __restrict__ A,
                            const _Float16* __restrict__ BT,
                            OutT* __restrict__ C, int M, int K, int Nout) {
  const int tilesN = Nout >> 4;
  const int tilesM = M >> 4;
  int wid = (blockIdx.x * blockDim.x + threadIdx.x) >> 5;
  int tm = wid / tilesN;
  int tn = wid - tm * tilesN;
  if (tm >= tilesM) return;            // whole-wave uniform exit: EXEC all-ones for WMMA
  int lane = threadIdx.x & 31;
  int l  = lane & 15;
  int hi = lane >> 4;

  const _Float16* arow = A  + (size_t)(tm * 16 + l) * K;  // A row M = tile+l
  const _Float16* bcol = BT + (size_t)(tn * 16 + l) * K;  // B col N = tile+l

  v8f c = {};
  for (int kk = 0; kk < K; kk += 32) {
    // A 16-bit 16x32 layout: lanes 0-15 -> K {kk..kk+7, kk+16..kk+23},
    //                        lanes 16-31 -> K {kk+8..kk+15, kk+24..kk+31}
    v8h alo = *(const v8h*)(arow + kk + hi * 8);
    v8h ahi = *(const v8h*)(arow + kk + hi * 8 + 16);
    // B 16-bit 32x16 layout: lanes 0-15 -> K {kk..kk+7, kk+16..kk+23},
    //                        lanes 16-31 -> K {kk+8..kk+15, kk+24..kk+31}
    v8h blo = *(const v8h*)(bcol + kk + hi * 8);
    v8h bhi = *(const v8h*)(bcol + kk + hi * 8 + 16);
    v16h a, b;
#pragma unroll
    for (int j = 0; j < 8; ++j) {
      a[j] = alo[j]; a[j + 8] = ahi[j];
      b[j] = blo[j]; b[j + 8] = bhi[j];
    }
    c = __builtin_amdgcn_wmma_f32_16x16x32_f16(false, a, false, b, (short)0, c,
                                               false, false);
  }
  // C layout: lane l (0-15) -> column n; VGPR r -> row r + hi*8
  int n = tn * 16 + l;
  size_t base = (size_t)(tm * 16 + hi * 8) * Nout + n;
#pragma unroll
  for (int r = 0; r < 8; ++r) C[base + (size_t)r * Nout] = (OutT)c[r];
}

// ---------------- weight transpose+convert: WT[o*Kpad+i] = W[rowOff+i][o] ----------------
__global__ void k_wT(const float* __restrict__ W, _Float16* __restrict__ WT,
                     int rowOff, int Ksl, int Kpad, int Nout, int ldW) {
  int tid = blockIdx.x * blockDim.x + threadIdx.x;
  if (tid >= Nout * Kpad) return;
  int o = tid / Kpad, i = tid - o * Kpad;
  WT[tid] = (i < Ksl) ? (_Float16)W[(size_t)(rowOff + i) * ldW + o] : (_Float16)0.f;
}

// embedding gather -> fp16 [N,128]
__global__ void k_emb(const int* __restrict__ xa, const float* __restrict__ emb,
                      _Float16* __restrict__ out) {
  int tid = blockIdx.x * blockDim.x + threadIdx.x;
  if (tid >= N_ * H_) return;
  int n = tid >> 7, c = tid & 127;
  out[tid] = (_Float16)emb[(size_t)xa[n] * H_ + c];
}

// edge_attr [E,50] f32 -> padded fp16 [E,64]
__global__ void k_eattr(const float* __restrict__ ea, _Float16* __restrict__ out) {
  int tid = blockIdx.x * blockDim.x + threadIdx.x;
  if (tid >= E_ * 64) return;
  int e = tid >> 6, c = tid & 63;
  out[tid] = (c < ED_) ? (_Float16)ea[(size_t)e * ED_ + c] : (_Float16)0.f;
}

// proj epilogue: h = silu(bn(t + b)); t (=aggr temp) zeroed for layer-0 scatter
__global__ void k_proj_post(float* __restrict__ t, const float* __restrict__ b,
                            const float* __restrict__ bnp, float* __restrict__ hf,
                            _Float16* __restrict__ hh) {
  int tid = blockIdx.x * blockDim.x + threadIdx.x;
  if (tid >= N_ * H_) return;
  int c = tid & 127;
  float x = siluf(bnf(t[tid] + b[c], bnp, H_, c));
  hf[tid] = x;
  hh[tid] = (_Float16)x;
  t[tid]  = 0.f;
}

// per-edge message + scatter-add. thread = (edge, 8 columns)
__global__ void k_edge(const int* __restrict__ ei,
                       const _Float16* __restrict__ Pdf, const _Float16* __restrict__ Psf,
                       const _Float16* __restrict__ Pds, const _Float16* __restrict__ Pss,
                       const _Float16* __restrict__ eAf, const _Float16* __restrict__ eAs,
                       const float* __restrict__ bf, const float* __restrict__ bs,
                       float* __restrict__ aggr) {
  int tid = blockIdx.x * blockDim.x + threadIdx.x;
  if (tid >= E_ * 16) return;
  int e  = tid >> 4;
  int c0 = (tid & 15) << 3;
  int s = ei[e], d = ei[E_ + e];
  v8h pdf = *(const v8h*)(Pdf + ((size_t)d << 7) + c0);
  v8h psf = *(const v8h*)(Psf + ((size_t)s << 7) + c0);
  v8h pds = *(const v8h*)(Pds + ((size_t)d << 7) + c0);
  v8h pss = *(const v8h*)(Pss + ((size_t)s << 7) + c0);
  v8h eaf = *(const v8h*)(eAf + ((size_t)e << 7) + c0);
  v8h eas = *(const v8h*)(eAs + ((size_t)e << 7) + c0);
  float* ag = aggr + ((size_t)d << 7) + c0;
#pragma unroll
  for (int j = 0; j < 8; ++j) {
    float lf = (float)pdf[j] + (float)psf[j] + (float)eaf[j] + bf[c0 + j];
    float ls = (float)pds[j] + (float)pss[j] + (float)eas[j] + bs[c0 + j];
    atomicAdd(ag + j, sigmoidf_(lf) * softplusf_(ls));
  }
}

// node update: h += silu(bn(h + aggr)); aggr re-zeroed for next layer
__global__ void k_node(float* __restrict__ hf, _Float16* __restrict__ hh,
                       float* __restrict__ aggr, const float* __restrict__ bnp) {
  int tid = blockIdx.x * blockDim.x + threadIdx.x;
  if (tid >= N_ * H_) return;
  int c = tid & 127;
  float t  = hf[tid] + aggr[tid];
  float nh = hf[tid] + siluf(bnf(t, bnp, H_, c));
  hf[tid]  = nh;
  hh[tid]  = (_Float16)nh;
  aggr[tid] = 0.f;
}

__global__ void k_init(float* __restrict__ gmax, float* __restrict__ den,
                       float* __restrict__ pooled) {
  int tid = blockIdx.x * blockDim.x + threadIdx.x;
  if (tid < NG_) gmax[tid] = -__builtin_inff();
  else if (tid < 2 * NG_) den[tid - NG_] = 0.f;
  else if (tid < 2 * NG_ + NG_ * H_) pooled[tid - 2 * NG_] = 0.f;
}

// gate scalar per node + float atomic max (ordered-int trick)
__global__ void k_gate(const float* __restrict__ G1, const float* __restrict__ b1,
                       const float* __restrict__ W2, const float* __restrict__ b2,
                       const int* __restrict__ batch, float* __restrict__ g,
                       float* __restrict__ gmax) {
  int n = blockIdx.x * blockDim.x + threadIdx.x;
  if (n >= N_) return;
  float acc = b2[0];
  const float* row = G1 + (size_t)n * 64;
#pragma unroll 8
  for (int j = 0; j < 64; ++j) acc += siluf(row[j] + b1[j]) * W2[j];
  g[n] = acc;
  int b = batch[n];
  if (acc >= 0.f) atomicMax((int*)(gmax + b), __float_as_int(acc));
  else            atomicMin((unsigned int*)(gmax + b), (unsigned int)__float_as_int(acc));
}

__global__ void k_expden(float* __restrict__ g, const float* __restrict__ gmax,
                         const int* __restrict__ batch, float* __restrict__ den) {
  int n = blockIdx.x * blockDim.x + threadIdx.x;
  if (n >= N_) return;
  int b = batch[n];
  float e = __expf(g[n] - gmax[b]);
  g[n] = e;
  atomicAdd(den + b, e);
}

__global__ void k_pool(const float* __restrict__ hf, const float* __restrict__ g,
                       const float* __restrict__ den, const int* __restrict__ batch,
                       float* __restrict__ pooled) {
  int tid = blockIdx.x * blockDim.x + threadIdx.x;
  if (tid >= N_ * H_) return;
  int n = tid >> 7, c = tid & 127;
  int b = batch[n];
  atomicAdd(pooled + ((size_t)b << 7) + c, hf[tid] * (g[n] / den[b]));
}

// tiny head MLP layer: act 0=none, 1=silu, 2=bn+silu
__global__ void k_mlp(const float* __restrict__ X, const float* __restrict__ W,
                      const float* __restrict__ b, const float* __restrict__ bnp,
                      float* __restrict__ Y, int In, int Out, int act) {
  int tid = blockIdx.x * blockDim.x + threadIdx.x;
  if (tid >= NG_ * Out) return;
  int gi = tid / Out, o = tid - gi * Out;
  float acc = b[o];
  const float* x = X + (size_t)gi * In;
  for (int i = 0; i < In; ++i) acc += x[i] * W[(size_t)i * Out + o];
  if (act == 2) acc = bnf(acc, bnp, Out, o);
  if (act)      acc = siluf(acc);
  Y[tid] = acc;
}

// ---------------- host ----------------
static inline int nb(long long t) { return (int)((t + 255) / 256); }

static inline void gemm16(const _Float16* A, const _Float16* BT, _Float16* C,
                          int M, int K, int Nout, hipStream_t s) {
  long long th = (long long)(M >> 4) * (Nout >> 4) * 32;
  k_gemm_wmma<_Float16><<<nb(th), 256, 0, s>>>(A, BT, C, M, K, Nout);
}
static inline void gemm32(const _Float16* A, const _Float16* BT, float* C,
                          int M, int K, int Nout, hipStream_t s) {
  long long th = (long long)(M >> 4) * (Nout >> 4) * 32;
  k_gemm_wmma<float><<<nb(th), 256, 0, s>>>(A, BT, C, M, K, Nout);
}

extern "C" void kernel_launch(void* const* d_in, const int* in_sizes, int n_in,
                              void* d_out, int out_size, void* d_ws, size_t ws_size,
                              hipStream_t stream) {
  (void)in_sizes; (void)n_in; (void)out_size; (void)ws_size;
  const int*   x_atom     = (const int*)d_in[0];
  const int*   edge_index = (const int*)d_in[1];
  const float* edge_attr  = (const float*)d_in[2];
  const int*   batch      = (const int*)d_in[3];
  const float* emb        = (const float*)d_in[4];
  const float* proj_W     = (const float*)d_in[5];
  const float* proj_b     = (const float*)d_in[6];
  const float* proj_bn    = (const float*)d_in[7];
  const float* conv_Wf    = (const float*)d_in[8];
  const float* conv_bf    = (const float*)d_in[9];
  const float* conv_Ws    = (const float*)d_in[10];
  const float* conv_bs    = (const float*)d_in[11];
  const float* conv_bn    = (const float*)d_in[12];
  const float* gate_W1    = (const float*)d_in[13];
  const float* gate_b1    = (const float*)d_in[14];
  const float* gate_W2    = (const float*)d_in[15];
  const float* gate_b2    = (const float*)d_in[16];
  const float* head_W1    = (const float*)d_in[17];
  const float* head_b1    = (const float*)d_in[18];
  const float* head_bn1   = (const float*)d_in[19];
  const float* head_W2    = (const float*)d_in[20];
  const float* head_b2    = (const float*)d_in[21];
  const float* head_bn2   = (const float*)d_in[22];
  const float* head_W3    = (const float*)d_in[23];
  const float* head_b3    = (const float*)d_in[24];
  const float* head_W4    = (const float*)d_in[25];
  const float* head_b4    = (const float*)d_in[26];
  float* out = (float*)d_out;

  // workspace carve-out
  char* ws = (char*)d_ws;
  auto carve = [&](size_t bytes) -> char* {
    char* p = ws; ws += (bytes + 255) & ~(size_t)255; return p;
  };
  float*     hf32    = (float*)carve((size_t)N_ * H_ * 4);
  _Float16*  hf16    = (_Float16*)carve((size_t)N_ * H_ * 2);
  _Float16*  bufA    = (_Float16*)carve((size_t)N_ * H_ * 2); // embf16, later G1 (N*64 f32)
  _Float16*  Pdf     = (_Float16*)carve((size_t)N_ * H_ * 2);
  _Float16*  Psf     = (_Float16*)carve((size_t)N_ * H_ * 2);
  _Float16*  Pds     = (_Float16*)carve((size_t)N_ * H_ * 2);
  _Float16*  Pss     = (_Float16*)carve((size_t)N_ * H_ * 2);
  _Float16*  eAf     = (_Float16*)carve((size_t)E_ * H_ * 2);
  _Float16*  eAs     = (_Float16*)carve((size_t)E_ * H_ * 2);
  float*     aggr    = (float*)carve((size_t)N_ * H_ * 4);
  _Float16*  eattr16 = (_Float16*)carve((size_t)E_ * 64 * 2);
  _Float16*  projWT  = (_Float16*)carve((size_t)H_ * H_ * 2);
  _Float16*  gateW1T = (_Float16*)carve((size_t)64 * H_ * 2);
  _Float16*  convT   = (_Float16*)carve((size_t)L_ * 81920 * 2);
  float*     g       = (float*)carve((size_t)N_ * 4);
  float*     gmax    = (float*)carve((size_t)NG_ * 4);
  float*     den     = (float*)carve((size_t)NG_ * 4);
  float*     pooled  = (float*)carve((size_t)NG_ * H_ * 4);
  float*     y1      = (float*)carve((size_t)NG_ * H_ * 4);
  float*     y2      = (float*)carve((size_t)NG_ * 64 * 4);
  float*     y3      = (float*)carve((size_t)NG_ * 32 * 4);

  // --- weight conversion (fp32 -> transposed fp16) ---
  k_wT<<<nb(128 * 128), 256, 0, stream>>>(proj_W, projWT, 0, 128, 128, 128, 128);
  k_wT<<<nb(64 * 128), 256, 0, stream>>>(gate_W1, gateW1T, 0, 128, 128, 64, 64);
  for (int l = 0; l < L_; ++l) {
    const float* Wf = conv_Wf + (size_t)l * Z_ * H_;
    const float* Wsm = conv_Ws + (size_t)l * Z_ * H_;
    _Float16* ct = convT + (size_t)l * 81920;
    k_wT<<<nb(16384), 256, 0, stream>>>(Wf,  ct + 0,     0,   128, 128, 128, 128); // Wf dst
    k_wT<<<nb(16384), 256, 0, stream>>>(Wf,  ct + 16384, 128, 128, 128, 128, 128); // Wf src
    k_wT<<<nb(16384), 256, 0, stream>>>(Wsm, ct + 32768, 0,   128, 128, 128, 128); // Ws dst
    k_wT<<<nb(16384), 256, 0, stream>>>(Wsm, ct + 49152, 128, 128, 128, 128, 128); // Ws src
    k_wT<<<nb(8192),  256, 0, stream>>>(Wf,  ct + 65536, 256, ED_, 64,  128, 128); // Wf edge
    k_wT<<<nb(8192),  256, 0, stream>>>(Wsm, ct + 73728, 256, ED_, 64,  128, 128); // Ws edge
  }
  k_eattr<<<nb((long long)E_ * 64), 256, 0, stream>>>(edge_attr, eattr16);
  k_emb<<<nb((long long)N_ * H_), 256, 0, stream>>>(x_atom, emb, bufA);

  // --- projection: h = silu(bn(emb[x] @ proj_W + b)) ---
  gemm32(bufA, projWT, aggr, N_, 128, 128, stream);
  k_proj_post<<<nb((long long)N_ * H_), 256, 0, stream>>>(aggr, proj_b, proj_bn, hf32, hf16);

  // --- 6 CGConv layers ---
  for (int l = 0; l < L_; ++l) {
    _Float16* ct = convT + (size_t)l * 81920;
    gemm16(hf16, ct + 0,     Pdf, N_, 128, 128, stream);
    gemm16(hf16, ct + 16384, Psf, N_, 128, 128, stream);
    gemm16(hf16, ct + 32768, Pds, N_, 128, 128, stream);
    gemm16(hf16, ct + 49152, Pss, N_, 128, 128, stream);
    gemm16(eattr16, ct + 65536, eAf, E_, 64, 128, stream);
    gemm16(eattr16, ct + 73728, eAs, E_, 64, 128, stream);
    k_edge<<<nb((long long)E_ * 16), 256, 0, stream>>>(
        edge_index, Pdf, Psf, Pds, Pss, eAf, eAs,
        conv_bf + (size_t)l * H_, conv_bs + (size_t)l * H_, aggr);
    k_node<<<nb((long long)N_ * H_), 256, 0, stream>>>(hf32, hf16, aggr,
                                                       conv_bn + (size_t)l * 4 * H_);
  }

  // --- global attention pooling ---
  float* G1 = (float*)bufA; // reuse (N*64 f32 == N*128 f16 bytes)
  gemm32(hf16, gateW1T, G1, N_, 128, 64, stream);
  k_init<<<nb(2 * NG_ + NG_ * H_), 256, 0, stream>>>(gmax, den, pooled);
  k_gate<<<nb(N_), 256, 0, stream>>>(G1, gate_b1, gate_W2, gate_b2, batch, g, gmax);
  k_expden<<<nb(N_), 256, 0, stream>>>(g, gmax, batch, den);
  k_pool<<<nb((long long)N_ * H_), 256, 0, stream>>>(hf32, g, den, batch, pooled);

  // --- head MLP ---
  k_mlp<<<nb(NG_ * 128), 256, 0, stream>>>(pooled, head_W1, head_b1, head_bn1, y1, 128, 128, 2);
  k_mlp<<<nb(NG_ * 64),  256, 0, stream>>>(y1, head_W2, head_b2, head_bn2, y2, 128, 64, 2);
  k_mlp<<<nb(NG_ * 32),  256, 0, stream>>>(y2, head_W3, head_b3, head_bn1, y3, 64, 32, 1);
  k_mlp<<<nb(NG_),       256, 0, stream>>>(y3, head_W4, head_b4, head_bn1, out, 32, 1, 0);
}
